// ScaledDotProductAttention_63960652972363
// MI455X (gfx1250) — compile-verified, matrix operand order checked by reading, not run on hardware
//
#include <hip/hip_runtime.h>
#include <hip/hip_bf16.h>

typedef __attribute__((ext_vector_type(16))) _Float16 v16h;
typedef __attribute__((ext_vector_type(8)))  _Float16 v8h;
typedef __attribute__((ext_vector_type(8)))  float    v8f;

#define S_LEN   2048
#define D_DIM   64
#define KSTEP   32
#define NSTEP   (S_LEN / KSTEP)   // 64
#define KPITCH  72                // halfs per K-stage row (64 + 8 pad -> conflict-free)
#define VPITCH  40                // halfs per Vt-stage row (32 + 8 pad)
#define PPITCH  40                // halfs per P-stage row
#define KTILE   (32 * KPITCH)     // halfs per K staging buffer
#define VTILE   (64 * VPITCH)     // halfs per Vt staging buffer

// D = A(16x32 f16) * B(32x16 f16) + C(16x16 f32)
#define WMMA(A, B, C) \
  __builtin_amdgcn_wmma_f32_16x16x32_f16(false, (A), false, (B), (short)0, (C), false, false)

// Build a 16x32 A/B fragment register (v16h): elems 0..7 at p[off], 8..15 at p[off+16]
static __device__ __forceinline__ v16h frag16(const _Float16* p, int off) {
  v8h a = *(const v8h*)(p + off);
  v8h b = *(const v8h*)(p + off + 16);
  return __builtin_shufflevector(a, b, 0,1,2,3,4,5,6,7,8,9,10,11,12,13,14,15);
}

// Cold (once-per-pass) reductions confined to 16-lane halves of the wave32.
static __device__ __forceinline__ float rowmax16(float v) {
  v = fmaxf(v, __shfl_xor(v, 1, 32));
  v = fmaxf(v, __shfl_xor(v, 2, 32));
  v = fmaxf(v, __shfl_xor(v, 4, 32));
  v = fmaxf(v, __shfl_xor(v, 8, 32));
  return v;
}
static __device__ __forceinline__ float rowsum16(float v) {
  v += __shfl_xor(v, 1, 32);
  v += __shfl_xor(v, 2, 32);
  v += __shfl_xor(v, 4, 32);
  v += __shfl_xor(v, 8, 32);
  return v;
}

// 8 contiguous fp32 values (two 128b global loads)
struct F8 { float4 a, b; };
static __device__ __forceinline__ F8 ld8(const float* p) {
  F8 r; r.a = *(const float4*)p; r.b = *(const float4*)(p + 4); return r;
}
// split fp32 -> f16 hi + f16 lo, contiguous store (K staging)
static __device__ __forceinline__ void cvt_store8(_Float16* dh, _Float16* dl, F8 f) {
  float x[8] = {f.a.x, f.a.y, f.a.z, f.a.w, f.b.x, f.b.y, f.b.z, f.b.w};
  v8h hv, lv;
  #pragma unroll
  for (int j = 0; j < 8; ++j) {
    _Float16 h = (_Float16)x[j];
    hv[j] = h; lv[j] = (_Float16)(x[j] - (float)h);
  }
  *(v8h*)dh = hv;
  *(v8h*)dl = lv;
}
// split fp32 -> f16 hi/lo, transposed scatter (V staging: [d][k])
static __device__ __forceinline__ void cvt_store_vt(_Float16* bh, _Float16* bl,
                                                    F8 f, int sc, int sr) {
  float y[8] = {f.a.x, f.a.y, f.a.z, f.a.w, f.b.x, f.b.y, f.b.z, f.b.w};
  #pragma unroll
  for (int j = 0; j < 8; ++j) {
    _Float16 h = (_Float16)y[j];
    bh[(sc + j) * VPITCH + sr] = h;
    bl[(sc + j) * VPITCH + sr] = (_Float16)(y[j] - (float)h);
  }
}

// S tile (16q x 16k) for key half `half` of the staged 32-key tile.
// 3-product hi/lo split over 2 d-chunks of 32 -> 6 WMMAs, fp32-class accuracy.
static __device__ __forceinline__ v8f score_tile(
    const _Float16* Kh, const _Float16* Kl, int half, int n, int off,
    v16h ah0, v16h ah1, v16h al0, v16h al1) {
  v8f acc = {};
  const _Float16* rh = Kh + (half * 16 + n) * KPITCH;
  const _Float16* rl = Kl + (half * 16 + n) * KPITCH;
  v16h bh0 = frag16(rh, off);
  v16h bl0 = frag16(rl, off);
  acc = WMMA(ah0, bh0, acc);
  acc = WMMA(ah0, bl0, acc);
  acc = WMMA(al0, bh0, acc);
  v16h bh1 = frag16(rh + 32, off);
  v16h bl1 = frag16(rl + 32, off);
  acc = WMMA(ah1, bh1, acc);
  acc = WMMA(ah1, bl1, acc);
  acc = WMMA(al1, bh1, acc);
  return acc;
}

__global__ __launch_bounds__(256)
void attn_fused_kernel(const float* __restrict__ Q,
                       const float* __restrict__ K,
                       const float* __restrict__ V,
                       float* __restrict__ ctx,
                       float* __restrict__ att) {
  __shared__ __align__(16) _Float16 sKh[2 * KTILE];     // double-buffered K hi
  __shared__ __align__(16) _Float16 sKl[2 * KTILE];     // double-buffered K lo
  __shared__ __align__(16) _Float16 sVh[2 * VTILE];     // double-buffered Vt hi
  __shared__ __align__(16) _Float16 sVl[2 * VTILE];     // double-buffered Vt lo
  __shared__ __align__(16) _Float16 sP[8][16 * PPITCH]; // per-wave P staging

  const int tid  = threadIdx.x;
  const int lane = tid & 31;
  const int wv   = tid >> 5;
  const int n    = lane & 15;       // N / column index within 16-lane half
  const int hi   = lane >> 4;       // which 16-lane half
  const int off  = hi ? 8 : 0;      // K-offset of this half in 16-bit A/B layout

  const int bh   = blockIdx.x >> 4; // (b*H + h), 0..63
  const int qblk = blockIdx.x & 15;
  const int q0   = qblk * 128 + wv * 16;  // this wave's first q row

  const float* Qb = Q + (size_t)bh * S_LEN * D_DIM;
  const float* Kb = K + (size_t)bh * S_LEN * D_DIM;
  const float* Vb = V + (size_t)bh * S_LEN * D_DIM;

  // ---- load this wave's 16x64 Q tile, fold 1/sqrt(64), split to f16 hi/lo A-frags
  v16h ah0, ah1, al0, al1;
  {
    const float* qr = Qb + (size_t)(q0 + n) * D_DIM;  // A-layout: row M = lane%16
    #pragma unroll
    for (int j = 0; j < 8; ++j) {
      float x0 = qr[off + j]       * 0.125f;
      float x1 = qr[off + 16 + j]  * 0.125f;
      float y0 = qr[32 + off + j]      * 0.125f;
      float y1 = qr[32 + off + 16 + j] * 0.125f;
      _Float16 h0 = (_Float16)x0, h1 = (_Float16)x1;
      _Float16 g0 = (_Float16)y0, g1 = (_Float16)y1;
      ah0[j] = h0; ah0[j + 8] = h1;
      ah1[j] = g0; ah1[j + 8] = g1;
      al0[j] = (_Float16)(x0 - (float)h0); al0[j + 8] = (_Float16)(x1 - (float)h1);
      al1[j] = (_Float16)(y0 - (float)g0); al1[j + 8] = (_Float16)(y1 - (float)g1);
    }
  }

  const int sr = tid >> 3;        // staged row 0..31 (cooperative K/V staging)
  const int sc = (tid & 7) * 8;   // staged column start
  const int kofs = sr * KPITCH + sc;

  // ========== Pass A: per-lane online softmax state (NO cross-lane in loop) ===
  float m[8], l[8];
  #pragma unroll
  for (int r = 0; r < 8; ++r) { m[r] = -3.0e38f; l[r] = 0.0f; }

  // prologue: stage tile 0 into buffer 0
  cvt_store8(sKh + kofs, sKl + kofs, ld8(Kb + (size_t)sr * D_DIM + sc));
  __syncthreads();

  for (int step = 0; step < NSTEP; ++step) {
    const int p  = step & 1;
    const int k0 = step * KSTEP;
    const bool more = (step + 1 < NSTEP);

    F8 knext;
    if (more) knext = ld8(Kb + (size_t)(k0 + KSTEP + sr) * D_DIM + sc);
    if (step + 2 < NSTEP)
      __builtin_prefetch(Kb + (size_t)(k0 + 2 * KSTEP + sr) * D_DIM + sc, 0, 1);

    const _Float16* kh = sKh + p * KTILE;
    const _Float16* kl = sKl + p * KTILE;
    v8f c0 = score_tile(kh, kl, 0, n, off, ah0, ah1, al0, al1);
    v8f c1 = score_tile(kh, kl, 1, n, off, ah0, ah1, al0, al1);

    #pragma unroll
    for (int r = 0; r < 8; ++r) {                 // pure per-lane recurrence
      float mn = fmaxf(m[r], fmaxf(c0[r], c1[r]));
      l[r] = l[r] * __expf(m[r] - mn) + __expf(c0[r] - mn) + __expf(c1[r] - mn);
      m[r] = mn;
    }

    if (more)
      cvt_store8(sKh + (p ^ 1) * KTILE + kofs, sKl + (p ^ 1) * KTILE + kofs, knext);
    __syncthreads();
  }

  // one cold cross-lane reduction per pass: merge 16 per-lane (m,l) pairs per row
  float rl_[8];
  #pragma unroll
  for (int r = 0; r < 8; ++r) {
    float mrow = rowmax16(m[r]);
    float lrow = rowsum16(l[r] * __expf(m[r] - mrow));
    m[r]   = mrow;
    rl_[r] = 1.0f / lrow;
  }

  // ====== Pass B: recompute S, write att = exp(s-m)/l, accumulate O += P*V ===
  v8f o[4] = {};
  float* attB = att + (size_t)bh * S_LEN * S_LEN;

  // prologue: stage K+V tile 0 into buffer 0
  cvt_store8(sKh + kofs, sKl + kofs, ld8(Kb + (size_t)sr * D_DIM + sc));
  cvt_store_vt(sVh, sVl, ld8(Vb + (size_t)sr * D_DIM + sc), sc, sr);
  __syncthreads();

  for (int step = 0; step < NSTEP; ++step) {
    const int p  = step & 1;
    const int k0 = step * KSTEP;
    const bool more = (step + 1 < NSTEP);

    F8 knext, vnext;
    if (more) {
      knext = ld8(Kb + (size_t)(k0 + KSTEP + sr) * D_DIM + sc);
      vnext = ld8(Vb + (size_t)(k0 + KSTEP + sr) * D_DIM + sc);
    }
    if (step + 2 < NSTEP) {
      __builtin_prefetch(Kb + (size_t)(k0 + 2 * KSTEP + sr) * D_DIM + sc, 0, 1);
      __builtin_prefetch(Vb + (size_t)(k0 + 2 * KSTEP + sr) * D_DIM + sc, 0, 1);
    }

    const _Float16* kh = sKh + p * KTILE;
    const _Float16* kl = sKl + p * KTILE;
    v8f c0 = score_tile(kh, kl, 0, n, off, ah0, ah1, al0, al1);
    v8f c1 = score_tile(kh, kl, 1, n, off, ah0, ah1, al0, al1);

    // normalize, write att, stage P (f16) for the A-matrix of the PV GEMM
    _Float16* prow = sP[wv];
    #pragma unroll
    for (int r = 0; r < 8; ++r) {
      float p0 = __expf(c0[r] - m[r]) * rl_[r];
      float p1 = __expf(c1[r] - m[r]) * rl_[r];
      const int q = q0 + 8 * hi + r;            // C-layout row for this lane
      float* ar = attB + (size_t)q * S_LEN + k0;
      ar[n]      = p0;
      ar[16 + n] = p1;
      prow[(r + 8 * hi) * PPITCH + n]      = (_Float16)p0;
      prow[(r + 8 * hi) * PPITCH + 16 + n] = (_Float16)p1;
    }
    __builtin_amdgcn_wave_barrier();  // DS ops are in-order per wave; keep order

    v16h pa = frag16(prow + n * PPITCH, off);   // A-layout: row = lane%16

    const _Float16* vh = sVh + p * VTILE;
    const _Float16* vl = sVl + p * VTILE;
    #pragma unroll
    for (int t = 0; t < 4; ++t) {               // O[16 x 64] in 4 N-tiles
      v16h bvh = frag16(vh + (16 * t + n) * VPITCH, off);
      v16h bvl = frag16(vl + (16 * t + n) * VPITCH, off);
      o[t] = WMMA(pa, bvh, o[t]);
      o[t] = WMMA(pa, bvl, o[t]);
    }

    if (more) {
      cvt_store8(sKh + (p ^ 1) * KTILE + kofs, sKl + (p ^ 1) * KTILE + kofs, knext);
      cvt_store_vt(sVh + (p ^ 1) * VTILE, sVl + (p ^ 1) * VTILE, vnext, sc, sr);
    }
    __syncthreads();
  }

  // ---- write context (already normalized since P was normalized)
  float* ctxB = ctx + (size_t)bh * S_LEN * D_DIM;
  #pragma unroll
  for (int t = 0; t < 4; ++t) {
    #pragma unroll
    for (int r = 0; r < 8; ++r) {
      const int q = q0 + 8 * hi + r;
      ctxB[(size_t)q * D_DIM + 16 * t + n] = o[t][r];
    }
  }
}

extern "C" void kernel_launch(void* const* d_in, const int* in_sizes, int n_in,
                              void* d_out, int out_size, void* d_ws, size_t ws_size,
                              hipStream_t stream) {
  (void)in_sizes; (void)n_in; (void)out_size; (void)d_ws; (void)ws_size;
  const float* Q = (const float*)d_in[0];
  const float* K = (const float*)d_in[1];
  const float* V = (const float*)d_in[2];
  float* out = (float*)d_out;
  float* ctx = out;                                      // [4,16,2048,64]
  float* att = out + (size_t)4 * 16 * 2048 * 64;         // [4,16,2048,2048]
  // 64 (b,h) pairs x 16 q-blocks of 128 rows; 256 threads = 8 wave32
  attn_fused_kernel<<<dim3(64 * 16), dim3(256), 0, stream>>>(Q, K, V, ctx, att);
}
